// Quanv3x3LayerClass_15470472200513
// MI455X (gfx1250) — compile-verified, matrix operand order checked by reading, not run on hardware
//
#include <hip/hip_runtime.h>
#include <hip/hip_bf16.h>

typedef __attribute__((ext_vector_type(16))) _Float16 v16h;
typedef __attribute__((ext_vector_type(8)))  _Float16 v8h;
typedef __attribute__((ext_vector_type(8)))  float    v8f;

#define NSIM   6272      // 8 * 28 * 28
#define NSTATE 512
#define NCH    4

// ---------------------------------------------------------------------------
// Kernel 0: zero the output (border stays zero; interior overwritten later)
// ---------------------------------------------------------------------------
__global__ void zero_out_kernel(float* __restrict__ out, int n) {
    int i = blockIdx.x * blockDim.x + threadIdx.x;
    if (i < n) out[i] = 0.0f;
}

// ---------------------------------------------------------------------------
// Kernel 1: build product states psi for all 6272 patches (f16 planes).
// psi_b = prod_q (bit_q ? sin h_q : cos h_q) * (-i)^popcount(b),
// h_q = pi*patch_q - pi/2, bit of qubit q at position (8-q).
// ---------------------------------------------------------------------------
__global__ void build_psi(const float* __restrict__ x,
                          _Float16* __restrict__ psiR,
                          _Float16* __restrict__ psiI) {
    const int s   = blockIdx.x;          // patch index
    const int b   = s / 784;
    const int rem = s % 784;
    const int r   = rem / 28;
    const int cc  = rem % 28;
    const int t   = threadIdx.x;         // 0..255

    __shared__ float cs[9], sn[9];
    if (t < 9) {
        const int dr = t / 3, dc = t % 3;
        const float* px = x + (((size_t)b * 30 + (r + dr)) * 30 + (cc + dc)) * 3;
        float avg = (px[0] + px[1] + px[2]) * (1.0f / 3.0f);
        float h = 3.14159265358979323846f * avg - 1.57079632679489661923f;
        cs[t] = __cosf(h);
        sn[t] = __sinf(h);
    }
    __syncthreads();

    for (int rep = 0; rep < 2; ++rep) {
        const int i = t + rep * 256;
        float mag = 1.0f;
#pragma unroll
        for (int q = 0; q < 9; ++q)
            mag *= ((i >> (8 - q)) & 1) ? sn[q] : cs[q];
        const int m = __popc(i) & 3;
        float re = 0.f, im = 0.f;
        if (m == 0)      re =  mag;
        else if (m == 1) im = -mag;
        else if (m == 2) re = -mag;
        else             im =  mag;
        psiR[(size_t)s * NSTATE + i] = (_Float16)re;
        psiI[(size_t)s * NSTATE + i] = (_Float16)im;
    }
}

// ---------------------------------------------------------------------------
// Kernel 2: build the per-channel 512x512 circuit unitary by propagating
// each basis state |k> through the 16 controlled gates (f32 in LDS).
// Stored row-major over n (uR[c][n][k]) so that the WMMA B-matrix
// (B[k][n] = U[n][k]) loads contiguously per lane.
// ---------------------------------------------------------------------------
__global__ void build_u(const float* __restrict__ qp,
                        _Float16* __restrict__ uR,
                        _Float16* __restrict__ uI) {
    const int ch = blockIdx.x >> 9;
    const int k  = blockIdx.x & 511;
    const int t  = threadIdx.x;          // 0..255

    __shared__ float2 st[NSTATE];
    st[t]       = make_float2(t == k ? 1.0f : 0.0f, 0.0f);
    st[t + 256] = make_float2((t + 256) == k ? 1.0f : 0.0f, 0.0f);
    __syncthreads();

    const float* th = qp + ch * 16;
    const int ctrlq[8] = {1, 3, 2, 8, 5, 7, 6, 4};
    const int tgtq[8]  = {0, 2, 0, 7, 4, 6, 4, 0};

#pragma unroll
    for (int g = 0; g < 8; ++g) {
        const int pc = 8 - ctrlq[g];     // ctrl bit position
        const int pt = 8 - tgtq[g];      // tgt  bit position
        const int low = t & ((1 << pt) - 1);
        const int i0  = ((t >> pt) << (pt + 1)) | low;
        const int i1  = i0 | (1 << pt);
        const bool act = ((i0 >> pc) & 1) != 0;

        { // controlled RZ(theta[2g]): diag(e^{-ih}, e^{+ih})
            float h = 0.5f * th[2 * g];
            float c = __cosf(h), sv = __sinf(h);
            if (act) {
                float2 a0 = st[i0], a1 = st[i1];
                st[i0] = make_float2(c * a0.x + sv * a0.y, c * a0.y - sv * a0.x);
                st[i1] = make_float2(c * a1.x - sv * a1.y, c * a1.y + sv * a1.x);
            }
        }
        __syncthreads();
        { // controlled RX(theta[2g+1]): [[c,-i s],[-i s,c]]
            float h = 0.5f * th[2 * g + 1];
            float c = __cosf(h), sv = __sinf(h);
            if (act) {
                float2 a0 = st[i0], a1 = st[i1];
                st[i0] = make_float2(c * a0.x + sv * a1.y,  c * a0.y - sv * a1.x);
                st[i1] = make_float2(sv * a0.y + c * a1.x, -sv * a0.x + c * a1.y);
            }
        }
        __syncthreads();
    }

    const size_t base = (size_t)ch * NSTATE * NSTATE;
    uR[base + (size_t)t * NSTATE + k]         = (_Float16)st[t].x;
    uI[base + (size_t)t * NSTATE + k]         = (_Float16)st[t].y;
    uR[base + (size_t)(t + 256) * NSTATE + k] = (_Float16)st[t + 256].x;
    uI[base + (size_t)(t + 256) * NSTATE + k] = (_Float16)st[t + 256].y;
}

// ---------------------------------------------------------------------------
// Kernel 3: complex GEMM phi = U * psi via v_wmma_f32_16x16x32_f16 with
// 2x2 (M,N) register tile blocking per wave. 2 accumulators per tile pair
// (phi_r via VALU-negated ai fragment), scalar-base + 32-bit-offset
// addressing, and __launch_bounds__(32,1) to lift the 128-VGPR cap that
// caused accumulator spills. Fused epilogue: z = sum_n sign(n)*|phi_n|^2.
// ---------------------------------------------------------------------------
__device__ __forceinline__ v16h load_frag(const char* base, int off) {
    v8h lo8 = *(const v8h*)(base + off);        // K = lo .. lo+7
    v8h hi8 = *(const v8h*)(base + off + 32);   // K = 16+lo .. 16+lo+7
    v16h r;
#pragma unroll
    for (int i = 0; i < 8; ++i) { r[i] = lo8[i]; r[8 + i] = hi8[i]; }
    return r;
}

__device__ __forceinline__ v16h neg_frag(v16h a) {
    v16h r;
#pragma unroll
    for (int i = 0; i < 16; ++i) r[i] = -a[i];   // v_xor 0x80008000 pairs
    return r;
}

#define WMMA_F16(A, B, C) \
    __builtin_amdgcn_wmma_f32_16x16x32_f16(false, (A), false, (B), (short)0, (C), false, false)

__global__ void __launch_bounds__(32, 1)
quanv_gemm_z(const _Float16* __restrict__ psiR,
             const _Float16* __restrict__ psiI,
             const _Float16* __restrict__ uR,
             const _Float16* __restrict__ uI,
             float* __restrict__ out) {
    const int mg     = blockIdx.x;            // 0..195 : 32-row block
    const int ch     = blockIdx.y;            // channel
    const int lane   = threadIdx.x;           // 0..31
    const int half16 = lane & 15;
    const int lo     = (lane < 16) ? 0 : 8;

    // Uniform scalar bases; per-lane 32-bit byte offsets (shared between the
    // real and imaginary planes, which have identical layout).
    const char* pAR = (const char*)psiR;
    const char* pAI = (const char*)psiI;
    const char* pBR = (const char*)(uR + (size_t)ch * NSTATE * NSTATE);
    const char* pBI = (const char*)(uI + (size_t)ch * NSTATE * NSTATE);

    const int offA0 = ((mg * 32 + half16) * NSTATE + lo) * 2;   // m-tile 0
    const int offA1 = offA0 + 16 * NSTATE * 2;                  // m-tile 1
    const int offBl = (half16 * NSTATE + lo) * 2;               // lane part of B

    float zacc0[8], zacc1[8];
#pragma unroll
    for (int j = 0; j < 8; ++j) { zacc0[j] = 0.0f; zacc1[j] = 0.0f; }

#pragma unroll 1
    for (int ng = 0; ng < 16; ++ng) {         // pairs of n-tiles {2ng, 2ng+1}
        const int offB0 = offBl + ng * 32 * NSTATE * 2;         // n-tile 2ng
        const int offB1 = offB0 + 16 * NSTATE * 2;              // n-tile 2ng+1

        // prefetch next n-group's B rows into cache (global_prefetch_b8)
        if (ng < 15) {
            __builtin_prefetch(pBR + offB0 + 32 * NSTATE * 2, 0, 1);
            __builtin_prefetch(pBI + offB0 + 32 * NSTATE * 2, 0, 1);
        }

        v8f accR00 = {}, accI00 = {};   // (m0,n0)
        v8f accR01 = {}, accI01 = {};   // (m0,n1)
        v8f accR10 = {}, accI10 = {};   // (m1,n0)
        v8f accR11 = {}, accI11 = {};   // (m1,n1)

#pragma unroll 1
        for (int kb = 0; kb < NSTATE * 2; kb += 64) {   // byte offset over K
            v16h ar0 = load_frag(pAR, offA0 + kb);
            v16h ai0 = load_frag(pAI, offA0 + kb);
            v16h ar1 = load_frag(pAR, offA1 + kb);
            v16h ai1 = load_frag(pAI, offA1 + kb);
            v16h br0 = load_frag(pBR, offB0 + kb);
            v16h bi0 = load_frag(pBI, offB0 + kb);
            v16h br1 = load_frag(pBR, offB1 + kb);
            v16h bi1 = load_frag(pBI, offB1 + kb);
            v16h an0 = neg_frag(ai0);   // f16 WMMA lacks A-neg modifier (ISA)
            v16h an1 = neg_frag(ai1);

            // phi_r += ar*br - ai*bi ; phi_i += ar*bi + ai*br
            accR00 = WMMA_F16(ar0, br0, accR00);
            accR00 = WMMA_F16(an0, bi0, accR00);
            accI00 = WMMA_F16(ar0, bi0, accI00);
            accI00 = WMMA_F16(ai0, br0, accI00);

            accR01 = WMMA_F16(ar0, br1, accR01);
            accR01 = WMMA_F16(an0, bi1, accR01);
            accI01 = WMMA_F16(ar0, bi1, accI01);
            accI01 = WMMA_F16(ai0, br1, accI01);

            accR10 = WMMA_F16(ar1, br0, accR10);
            accR10 = WMMA_F16(an1, bi0, accR10);
            accI10 = WMMA_F16(ar1, bi0, accI10);
            accI10 = WMMA_F16(ai1, br0, accI10);

            accR11 = WMMA_F16(ar1, br1, accR11);
            accR11 = WMMA_F16(an1, bi1, accR11);
            accI11 = WMMA_F16(ar1, bi1, accI11);
            accI11 = WMMA_F16(ai1, br1, accI11);
        }

        // sign(n): qubit 0 is basis-index MSB -> +1 for n<256; tiles 2ng and
        // 2ng+1 share the sign (boundary is at tile 16, ng<8 <=> n<256).
        const float sgn = (ng < 8) ? 1.0f : -1.0f;
#pragma unroll
        for (int j = 0; j < 8; ++j) {
            zacc0[j] += sgn * (accR00[j] * accR00[j] + accI00[j] * accI00[j]
                             + accR01[j] * accR01[j] + accI01[j] * accI01[j]);
            zacc1[j] += sgn * (accR10[j] * accR10[j] + accI10[j] * accI10[j]
                             + accR11[j] * accR11[j] + accI11[j] * accI11[j]);
        }
    }

    // Reduce over the N dimension (striped across lanes) per output row.
    // C/D layout: lane l, VGPR j holds element (M = j + (l<16 ? 0 : 8), N = l&15).
    __shared__ float zsh[32][8];
#pragma unroll 1
    for (int mt = 0; mt < 2; ++mt) {
#pragma unroll
        for (int j = 0; j < 8; ++j) zsh[lane][j] = (mt == 0) ? zacc0[j] : zacc1[j];
        __syncthreads();
        if (lane < 16) {
            const int base = (lane < 8) ? 0 : 16;
            const int j    = lane & 7;
            float sum = 0.0f;
#pragma unroll
            for (int l = 0; l < 16; ++l) sum += zsh[base + l][j];

            const int s   = mg * 32 + mt * 16 + lane;
            const int b   = s / 784;
            const int rem = s % 784;
            const int r   = rem / 28;
            const int cc  = rem % 28;
            out[(((size_t)b * 30 + (r + 1)) * 30 + (cc + 1)) * NCH + ch] =
                (sum + 1.0f) * 0.5f;
        }
        __syncthreads();
    }
}

// ---------------------------------------------------------------------------
// Launch
// ---------------------------------------------------------------------------
extern "C" void kernel_launch(void* const* d_in, const int* in_sizes, int n_in,
                              void* d_out, int out_size, void* d_ws, size_t ws_size,
                              hipStream_t stream) {
    const float* x  = (const float*)d_in[0];   // (8,30,30,3) f32
    const float* qp = (const float*)d_in[1];   // (4,16) f32
    float* out = (float*)d_out;                // (8,30,30,4) f32

    char* ws = (char*)d_ws;
    const size_t PSI_BYTES = (size_t)NSIM * NSTATE * sizeof(_Float16);        // 6,422,528
    const size_t U_BYTES   = (size_t)NCH * NSTATE * NSTATE * sizeof(_Float16);// 2,097,152
    _Float16* psiR = (_Float16*)(ws);
    _Float16* psiI = (_Float16*)(ws + PSI_BYTES);
    _Float16* uR   = (_Float16*)(ws + 2 * PSI_BYTES);
    _Float16* uI   = (_Float16*)(ws + 2 * PSI_BYTES + U_BYTES);

    zero_out_kernel<<<(out_size + 255) / 256, 256, 0, stream>>>(out, out_size);
    build_psi<<<NSIM, 256, 0, stream>>>(x, psiR, psiI);
    build_u<<<NCH * NSTATE, 256, 0, stream>>>(qp, uR, uI);
    quanv_gemm_z<<<dim3(NSIM / 32, NCH), 32, 0, stream>>>(psiR, psiI, uR, uI, out);
}
